// LSTM_22428319220552
// MI455X (gfx1250) — matrix operand order repaired and checked by reading, not compiled, not run on hardware
//
#include <hip/hip_runtime.h>

#define T_LEN 2048
#define BATCH 4
#define ISZ   1024
#define HSZ   1024
#define G4    (4 * HSZ)
#define NWG   256          // persistent workgroups in the scan kernel; each owns 4 hidden units

typedef __attribute__((ext_vector_type(16))) __bf16 bf16x16;
typedef __attribute__((ext_vector_type(8)))  float  f32x8;

union Frag { bf16x16 v; uint4 q[2]; };

__device__ __forceinline__ unsigned short f2bf(float f) {
  unsigned u = __float_as_uint(f);
  u += 0x7FFFu + ((u >> 16) & 1u);            // round-to-nearest-even
  return (unsigned short)(u >> 16);
}
__device__ __forceinline__ float bf2f(unsigned short b) {
  return __uint_as_float(((unsigned)b) << 16);
}

// ---------------------------------------------------------------- fp32 -> bf16
__global__ void __launch_bounds__(256) cvt_f32_bf16(const float* __restrict__ src,
                                                    unsigned short* __restrict__ dst,
                                                    int n4) {
  int i = blockIdx.x * blockDim.x + threadIdx.x;
  if (i >= n4) return;
  float4 f = reinterpret_cast<const float4*>(src)[i];
  ushort4 o;
  o.x = f2bf(f.x); o.y = f2bf(f.y); o.z = f2bf(f.z); o.w = f2bf(f.w);
  reinterpret_cast<ushort4*>(dst)[i] = o;
}

// ------------------------------------------------- zero h ping-pong + barrier
__global__ void __launch_bounds__(256) init_state(unsigned short* __restrict__ hbuf,
                                                  unsigned* __restrict__ bar) {
  int i = blockIdx.x * blockDim.x + threadIdx.x;
  if (i < 2 * 4 * HSZ) hbuf[i] = 0;           // both h buffers (rows b=0..3 only)
  if (i < 2) bar[i] = 0;
}

// --------------------------------------------------------- pre = x @ W_in.T
// A = x_bf16 [8192 x 1024] (K contig), B rows = W_in rows (K contig).
// Register blocking: each wave computes 4 adjacent 16x16 column tiles, sharing
// one A fragment per k-step. Distinct B fragments keep ~10 loads in flight
// behind each v_wmma_f32_16x16x32_bf16 (no full loadcnt drains).
__global__ void __launch_bounds__(256) gemm_pre(const unsigned short* __restrict__ X,
                                                const unsigned short* __restrict__ W,
                                                unsigned short* __restrict__ P) {
  const int lane = threadIdx.x & 31;
  const int wave = threadIdx.x >> 5;
  const int wid  = blockIdx.x * 8 + wave;     // 32768 wave-jobs
  const int rowTile  = wid >> 6;              // 0..511
  const int colGroup = wid & 63;              // 0..63  (4 col tiles each)
  const int r   = lane & 15;
  const int sel = lane >> 4;

  const unsigned short* arow  = X + (size_t)(rowTile * 16 + r) * ISZ;
  const unsigned short* brow0 = W + (size_t)(colGroup * 64 + 0 * 16 + r) * ISZ;
  const unsigned short* brow1 = W + (size_t)(colGroup * 64 + 1 * 16 + r) * ISZ;
  const unsigned short* brow2 = W + (size_t)(colGroup * 64 + 2 * 16 + r) * ISZ;
  const unsigned short* brow3 = W + (size_t)(colGroup * 64 + 3 * 16 + r) * ISZ;

  f32x8 c0 = {}, c1 = {}, c2 = {}, c3 = {};
  for (int kb = 0; kb < 32; ++kb) {
    const int K0 = kb * 32;
    Frag a, b0, b1, b2, b3;
    a.q[0]  = *reinterpret_cast<const uint4*>(arow  + K0 + sel * 16);
    a.q[1]  = *reinterpret_cast<const uint4*>(arow  + K0 + sel * 16 + 8);
    b0.q[0] = *reinterpret_cast<const uint4*>(brow0 + K0 + sel * 16);
    b0.q[1] = *reinterpret_cast<const uint4*>(brow0 + K0 + sel * 16 + 8);
    b1.q[0] = *reinterpret_cast<const uint4*>(brow1 + K0 + sel * 16);
    b1.q[1] = *reinterpret_cast<const uint4*>(brow1 + K0 + sel * 16 + 8);
    b2.q[0] = *reinterpret_cast<const uint4*>(brow2 + K0 + sel * 16);
    b2.q[1] = *reinterpret_cast<const uint4*>(brow2 + K0 + sel * 16 + 8);
    b3.q[0] = *reinterpret_cast<const uint4*>(brow3 + K0 + sel * 16);
    b3.q[1] = *reinterpret_cast<const uint4*>(brow3 + K0 + sel * 16 + 8);
    c0 = __builtin_amdgcn_wmma_f32_16x16x32_bf16(false, a.v, false, b0.v, (short)0, c0, false, false);
    c1 = __builtin_amdgcn_wmma_f32_16x16x32_bf16(false, a.v, false, b1.v, (short)0, c1, false, false);
    c2 = __builtin_amdgcn_wmma_f32_16x16x32_bf16(false, a.v, false, b2.v, (short)0, c2, false, false);
    c3 = __builtin_amdgcn_wmma_f32_16x16x32_bf16(false, a.v, false, b3.v, (short)0, c3, false, false);
  }
#pragma unroll
  for (int v = 0; v < 8; ++v) {
    int row = rowTile * 16 + v + sel * 8;
    unsigned short* prow = P + (size_t)row * G4 + colGroup * 64 + r;
    prow[0]  = f2bf(c0[v]);
    prow[16] = f2bf(c1[v]);
    prow[32] = f2bf(c2[v]);
    prow[48] = f2bf(c3[v]);
  }
}

// ------------------------------------------------------------- LSTM scan
// WG g owns hidden units [g*4, g*4+4). Its 16 B-matrix rows (4 gates x 4 units)
// of W_hh live in LDS (32KB) for the whole kernel -> entire W_hh is LDS-resident
// across the 256 WGs. Waves split K=1024 four ways (8 WMMAs each), A rows 0..3
// read straight from the L2-broadcast h buffer, partials reduced in LDS, fp32
// cell state in registers, one grid barrier per step.
__global__ void __launch_bounds__(128) lstm_scan(const unsigned short* __restrict__ Whh,
                                                 const unsigned short* __restrict__ Pre,
                                                 unsigned short* __restrict__ hbuf, // [2][4][HSZ]
                                                 float* __restrict__ out,
                                                 float* __restrict__ hT,
                                                 float* __restrict__ cT,
                                                 unsigned* __restrict__ bar) {
  __shared__ unsigned short sW[16 * HSZ];   // 32KB: B rows n=0..15 (n = gate*4 + unit)
  __shared__ float sPart[4 * 4 * 16];       // [wave][b][n] partial sums
  __shared__ float sG[4 * 16];              // [b][n] full gate pre-activations

  const int tid  = threadIdx.x;
  const int lane = tid & 31;
  const int wave = tid >> 5;
  const int wg   = blockIdx.x;

  // stage W_hh slice: LDS row n  <-  W_hh row (n>>2)*HSZ + wg*4 + (n&3)
  for (int idx = tid; idx < 16 * (HSZ / 8); idx += 128) {
    int n  = idx >> 7;                      // 128 x uint4 per row
    int ch = idx & 127;
    int grow = (n >> 2) * HSZ + wg * 4 + (n & 3);
    reinterpret_cast<uint4*>(sW)[n * (HSZ / 8) + ch] =
        reinterpret_cast<const uint4*>(Whh + (size_t)grow * HSZ)[ch];
  }
  __syncthreads();

  const int r   = lane & 15;
  const int sel = lane >> 4;
  const int K0w = wave * 256;
  const int eb = tid >> 2, eu = tid & 3;    // elementwise mapping for tid < 16
  const int pb = tid >> 4, pn = tid & 15;   // pre/reduce mapping for tid < 64
  float cst = 0.0f, hlast = 0.0f;
  unsigned genTarget = 0;

  for (int t = 0; t < T_LEN; ++t) {
    const unsigned short* hin  = hbuf + (size_t)(t & 1) * 4 * HSZ;
    unsigned short*       hout = hbuf + (size_t)((t + 1) & 1) * 4 * HSZ;

    // warm L2/L0 for next step's scattered pre gather while WMMAs run
    if (tid < 64 && t + 1 < T_LEN)
      __builtin_prefetch(
          &Pre[(size_t)((t + 1) * BATCH + pb) * G4 + (pn >> 2) * HSZ + wg * 4 + (pn & 3)], 0, 3);

    f32x8 c = {};
#pragma unroll
    for (int kbl = 0; kbl < 8; ++kbl) {
      const int K0 = K0w + kbl * 32;
      Frag a, b;
      if (r < 4) {                          // only batch rows 0..3 exist; rest are zero
        a.q[0] = *reinterpret_cast<const uint4*>(hin + r * HSZ + K0 + sel * 16);
        a.q[1] = *reinterpret_cast<const uint4*>(hin + r * HSZ + K0 + sel * 16 + 8);
      } else {
        a.q[0] = make_uint4(0, 0, 0, 0);
        a.q[1] = make_uint4(0, 0, 0, 0);
      }
      b.q[0] = *reinterpret_cast<const uint4*>(sW + r * HSZ + K0 + sel * 16);
      b.q[1] = *reinterpret_cast<const uint4*>(sW + r * HSZ + K0 + sel * 16 + 8);
      c = __builtin_amdgcn_wmma_f32_16x16x32_bf16(false, a.v, false, b.v,
                                                  (short)0, c, false, false);
    }
    if (lane < 16) {
#pragma unroll
      for (int v = 0; v < 4; ++v) sPart[(wave * 4 + v) * 16 + lane] = c[v];
    }
    __syncthreads();

    if (tid < 64) {                         // reduce K-partials + add precomputed input GEMM
      float g = bf2f(Pre[(size_t)(t * BATCH + pb) * G4 + (pn >> 2) * HSZ + wg * 4 + (pn & 3)]);
      g += sPart[(0 * 4 + pb) * 16 + pn] + sPart[(1 * 4 + pb) * 16 + pn] +
           sPart[(2 * 4 + pb) * 16 + pn] + sPart[(3 * 4 + pb) * 16 + pn];
      sG[pb * 16 + pn] = g;
    }
    __syncthreads();

    if (tid < 16) {                         // gate nonlinearities + state update
      float gi = sG[eb * 16 + 0  + eu];
      float gf = sG[eb * 16 + 4  + eu];
      float gg = sG[eb * 16 + 8  + eu];
      float go = sG[eb * 16 + 12 + eu];
      float si = 1.0f / (1.0f + __expf(-gi));
      float sf = 1.0f / (1.0f + __expf(-gf));
      float so = 1.0f / (1.0f + __expf(-go));
      float cn = sf * cst + si * tanhf(gg);
      float hn = so * tanhf(cn);
      cst = cn; hlast = hn;
      int col = wg * 4 + eu;
      out[(size_t)t * (BATCH * HSZ) + eb * HSZ + col] = hn;
      hout[eb * HSZ + col] = f2bf(hn);
    }
    __threadfence();
    __syncthreads();

    // grid-wide sense barrier (count, generation) over NWG persistent WGs
    ++genTarget;
    if (tid == 0) {
      unsigned old = __hip_atomic_fetch_add(&bar[0], 1u, __ATOMIC_ACQ_REL,
                                            __HIP_MEMORY_SCOPE_AGENT);
      if (old == NWG - 1) {
        __hip_atomic_store(&bar[0], 0u, __ATOMIC_RELAXED, __HIP_MEMORY_SCOPE_AGENT);
        __hip_atomic_fetch_add(&bar[1], 1u, __ATOMIC_RELEASE, __HIP_MEMORY_SCOPE_AGENT);
      } else {
        while (__hip_atomic_load(&bar[1], __ATOMIC_ACQUIRE,
                                 __HIP_MEMORY_SCOPE_AGENT) < genTarget) {
          __builtin_amdgcn_s_sleep(1);
        }
      }
    }
    __syncthreads();
  }

  if (tid < 16) {
    int col = wg * 4 + eu;
    hT[eb * HSZ + col] = hlast;
    cT[eb * HSZ + col] = cst;
  }
}

// ---------------------------------------------------------------------------
extern "C" void kernel_launch(void* const* d_in, const int* in_sizes, int n_in,
                              void* d_out, int out_size, void* d_ws, size_t ws_size,
                              hipStream_t stream) {
  const float* x    = (const float*)d_in[0];   // [T,B,I]
  const float* W_in = (const float*)d_in[1];   // [4H,I]
  const float* W_hh = (const float*)d_in[2];   // [4H,H]

  // workspace layout (bf16 stagings + pre + h ping-pong + barrier)
  unsigned short* xb   = (unsigned short*)d_ws;                 // 8,388,608
  unsigned short* wib  = xb  + (size_t)T_LEN * BATCH * ISZ;     // 4,194,304
  unsigned short* whb  = wib + (size_t)G4 * ISZ;                // 4,194,304
  unsigned short* pre  = whb + (size_t)G4 * HSZ;                // 33,554,432
  unsigned short* hbuf = pre + (size_t)T_LEN * BATCH * G4;      // 8,192 (2 x 4 x H)
  unsigned*       bar  = (unsigned*)(hbuf + 2 * 4 * HSZ);

  size_t need = (size_t)((unsigned short*)(bar + 2) - (unsigned short*)d_ws) * sizeof(unsigned short);
  if (ws_size < need) return;

  float* out = (float*)d_out;                       // [T,B,H]
  float* hT  = out + (size_t)T_LEN * BATCH * HSZ;   // [B,H]
  float* cT  = hT + (size_t)BATCH * HSZ;            // [B,H]

  // 1) convert operands to bf16
  cvt_f32_bf16<<<(T_LEN * BATCH * ISZ / 4 + 255) / 256, 256, 0, stream>>>(x, xb, T_LEN * BATCH * ISZ / 4);
  cvt_f32_bf16<<<(G4 * ISZ / 4 + 255) / 256, 256, 0, stream>>>(W_in, wib, G4 * ISZ / 4);
  cvt_f32_bf16<<<(G4 * HSZ / 4 + 255) / 256, 256, 0, stream>>>(W_hh, whb, G4 * HSZ / 4);

  // 2) zero h ping-pong buffers + grid barrier state (every call: determinism)
  init_state<<<(2 * 4 * HSZ + 255) / 256, 256, 0, stream>>>(hbuf, bar);

  // 3) pre = x @ W_in.T   (512 row-tiles x 64 col-groups, 4 tiles per wave)
  gemm_pre<<<(512 * 64) / 8, 256, 0, stream>>>(xb, wib, pre);

  // 4) persistent recurrent scan, W_hh LDS-resident across all 2048 steps
  lstm_scan<<<NWG, 128, 0, stream>>>(whb, pre, hbuf, out, hT, cT, bar);
}
